// Model_73856257622085
// MI455X (gfx1250) — compile-verified, matrix operand order checked by reading, not run on hardware
//
#include <hip/hip_runtime.h>
#include <math.h>

typedef _Float16 h16;
typedef __attribute__((ext_vector_type(16))) _Float16 v16h;
typedef __attribute__((ext_vector_type(8)))  _Float16 h8;
typedef __attribute__((ext_vector_type(8)))  float    v8f;

#define NQn 100000
#define NAn 200000
#define E1n 250000
#define E2n 250000
#define E3n 500000
#define NLn 100000

union H16x16 { v16h v; h16 s[16]; h8 h2[2]; };

__device__ __forceinline__ v8f wmma16(v16h a, v16h b, v8f c) {
  // D = A(16x32 f16) * B(32x16 f16) + C(16x16 f32)
  return __builtin_amdgcn_wmma_f32_16x16x32_f16(false, a, false, b, (short)0, c, false, false);
}

__device__ __forceinline__ float gelu_t(float x) {
  return 0.5f * x * (1.0f + tanhf(0.7978845608028654f * (x + 0.044715f * x * x * x)));
}

// A-fragment from row-major f32: lane holds row (lane&15); elems 0..7 -> K=k0+half*8+j,
// elems 8..15 -> K=k0+16+half*8+j   (ISA 16-bit A 16x32 layout)
template<int ACT>
__device__ __forceinline__ v16h load_afrag(const float* __restrict__ A, long row0, int lda,
                                           int k0, int lane) {
  int r = lane & 15, half = lane >> 4;
  const float* p = A + (row0 + r) * (long)lda + k0 + half * 8;
  H16x16 u;
#pragma unroll
  for (int j = 0; j < 8; ++j) { float v = p[j];      if (ACT) v = gelu_t(v); u.s[j]     = (h16)v; }
#pragma unroll
  for (int j = 0; j < 8; ++j) { float v = p[16 + j]; if (ACT) v = gelu_t(v); u.s[8 + j] = (h16)v; }
  return u.v;
}

// B-fragment from pre-packed [ks][nt][lane][16] f16 buffer: fully coalesced
__device__ __forceinline__ v16h load_bfrag(const h16* __restrict__ P, int ks, int nt, int lane) {
  const h8* q = (const h8*)(P + (((ks * 4 + nt) * 32 + lane) * 16));
  H16x16 u; u.h2[0] = q[0]; u.h2[1] = q[1];
  return u.v;
}

// A-fragment from an LDS-resident [16][64] f16 tile (C->A layout re-striping)
__device__ __forceinline__ v16h lds_afrag(const h16* T, int ks, int lane) {
  int r = lane & 15, half = lane >> 4;
  const h16* p = T + r * 64 + ks * 32 + half * 8;
  H16x16 u; u.h2[0] = *(const h8*)p; u.h2[1] = *(const h8*)(p + 16);
  return u.v;
}

__device__ __forceinline__ void atomicMaxF(float* addr, float val) {
  if (val >= 0.f) atomicMax((int*)addr, __float_as_int(val));
  else            atomicMin((unsigned int*)addr, __float_as_uint(val));
}

// ---------------- weight packing into WMMA B-fragment order ----------------
__global__ void k_packB(const float* __restrict__ W, int ksteps, h16* __restrict__ out) {
  int idx = blockIdx.x * 256 + threadIdx.x;
  int total = ksteps * 2048;               // ksteps*4nt*32lane*16
  if (idx >= total) return;
  int j = idx & 15, lane = (idx >> 4) & 31, nt = (idx >> 9) & 3, ks = idx >> 11;
  int k = ks * 32 + (lane >> 4) * 8 + (j & 7) + ((j >> 3) << 4);
  int n = nt * 16 + (lane & 15);
  out[idx] = (h16)W[k * 64 + n];
}

// a_rel/m_rel [3][2][32][32] -> per relation t: 4 nt-frags (head = nt>>1, K=in-head dim)
__global__ void k_packRel(const float* __restrict__ R, h16* __restrict__ out) {
  int idx = blockIdx.x * 256 + threadIdx.x;
  if (idx >= 3 * 2048) return;
  int j = idx & 15, lane = (idx >> 4) & 31, nt = (idx >> 9) & 3, t = idx >> 11;
  int k = (lane >> 4) * 8 + (j & 7) + ((j >> 3) << 4);
  int hh = nt >> 1;
  int n = (nt & 1) * 16 + (lane & 15);
  out[idx] = (h16)R[((t * 2 + hh) * 32 + k) * 32 + n];
}

__global__ void k_fill(float* __restrict__ p, int n, float v) {
  int i = blockIdx.x * blockDim.x + threadIdx.x;
  if (i < n) p[i] = v;
}

// ---------------- stage A: h = relu(x @ W_in + b), K=128 ----------------
__global__ void k_gemm_in(const float* __restrict__ X, int ntiles,
                          const h16* __restrict__ Bp, const float* __restrict__ bias,
                          float* __restrict__ H) {
  int gw = (blockIdx.x * blockDim.x + threadIdx.x) >> 5;
  int lane = threadIdx.x & 31;
  if (gw >= ntiles) return;
  long row0 = (long)gw * 16;
  int nlo = lane & 15, half = lane >> 4;
  const v8f zero = {0.f,0.f,0.f,0.f,0.f,0.f,0.f,0.f};
  v8f acc[4];
#pragma unroll
  for (int nt = 0; nt < 4; ++nt) acc[nt] = zero;
#pragma unroll
  for (int ks = 0; ks < 4; ++ks) {
    v16h a = load_afrag<0>(X, row0, 128, ks * 32, lane);
#pragma unroll
    for (int nt = 0; nt < 4; ++nt) acc[nt] = wmma16(a, load_bfrag(Bp, ks, nt, lane), acc[nt]);
  }
#pragma unroll
  for (int nt = 0; nt < 4; ++nt) {
    int n = nt * 16 + nlo; float b = bias[n];
#pragma unroll
    for (int i = 0; i < 8; ++i) {
      float v = acc[nt][i] + b;
      H[(row0 + i + half * 8) * 64 + n] = v > 0.f ? v : 0.f;
    }
  }
}

// ------- stage B+C fused (question nodes): q, and kr0/kr1/vr0/vr1 via LDS re-striping ------
__global__ void k_qkv_rel_q(const float* __restrict__ Hin, int ntiles,
    const h16* __restrict__ pWk, const h16* __restrict__ pWq, const h16* __restrict__ pWv,
    const float* __restrict__ bk, const float* __restrict__ bq, const float* __restrict__ bv,
    const h16* __restrict__ pA, const h16* __restrict__ pM,
    float* __restrict__ Q, float* __restrict__ kr0, float* __restrict__ kr1,
    float* __restrict__ vr0, float* __restrict__ vr1) {
  __shared__ h16 kt[8][1024];
  __shared__ h16 vt[8][1024];
  int wv = threadIdx.x >> 5, lane = threadIdx.x & 31;
  int tile = blockIdx.x * 8 + wv;
  if (tile >= ntiles) tile = ntiles - 1;        // clamp (dup work) so barrier is uniform
  long row0 = (long)tile * 16;
  int nlo = lane & 15, half = lane >> 4;
  const v8f zero = {0.f,0.f,0.f,0.f,0.f,0.f,0.f,0.f};
  v16h a0 = load_afrag<0>(Hin, row0, 64, 0, lane);
  v16h a1 = load_afrag<0>(Hin, row0, 64, 32, lane);
#pragma unroll
  for (int nt = 0; nt < 4; ++nt) {
    int n = nt * 16 + nlo;
    v8f aq = wmma16(a0, load_bfrag(pWq, 0, nt, lane), zero);
    aq = wmma16(a1, load_bfrag(pWq, 1, nt, lane), aq);
    float b = bq[n];
#pragma unroll
    for (int i = 0; i < 8; ++i) Q[(row0 + i + half * 8) * 64 + n] = aq[i] + b;
    v8f ak = wmma16(a0, load_bfrag(pWk, 0, nt, lane), zero);
    ak = wmma16(a1, load_bfrag(pWk, 1, nt, lane), ak);
    float bb = bk[n];
#pragma unroll
    for (int i = 0; i < 8; ++i) kt[wv][(i + half * 8) * 64 + n] = (h16)(ak[i] + bb);
    v8f av = wmma16(a0, load_bfrag(pWv, 0, nt, lane), zero);
    av = wmma16(a1, load_bfrag(pWv, 1, nt, lane), av);
    float bv2 = bv[n];
#pragma unroll
    for (int i = 0; i < 8; ++i) vt[wv][(i + half * 8) * 64 + n] = (h16)(av[i] + bv2);
  }
  __syncthreads();
  v16h ka[2] = { lds_afrag(kt[wv], 0, lane), lds_afrag(kt[wv], 1, lane) };
  v16h va[2] = { lds_afrag(vt[wv], 0, lane), lds_afrag(vt[wv], 1, lane) };
#pragma unroll
  for (int t = 0; t < 2; ++t) {
    float* ko = t ? kr1 : kr0;
    float* vo = t ? vr1 : vr0;
#pragma unroll
    for (int nt = 0; nt < 4; ++nt) {         // head = nt>>1, single K=32 step per head
      int n = nt * 16 + nlo;
      v8f ac = wmma16(ka[nt >> 1], load_bfrag(pA, t, nt, lane), zero);
#pragma unroll
      for (int i = 0; i < 8; ++i) ko[(row0 + i + half * 8) * 64 + n] = ac[i];
      v8f am = wmma16(va[nt >> 1], load_bfrag(pM, t, nt, lane), zero);
#pragma unroll
      for (int i = 0; i < 8; ++i) vo[(row0 + i + half * 8) * 64 + n] = am[i];
    }
  }
}

// ------- stage B+C fused (answer nodes): q, kr2, vr2 -------
__global__ void k_qkv_rel_a(const float* __restrict__ Hin, int ntiles,
    const h16* __restrict__ pWk, const h16* __restrict__ pWq, const h16* __restrict__ pWv,
    const float* __restrict__ bk, const float* __restrict__ bq, const float* __restrict__ bv,
    const h16* __restrict__ pA, const h16* __restrict__ pM,
    float* __restrict__ Q, float* __restrict__ kr2, float* __restrict__ vr2) {
  __shared__ h16 kt[8][1024];
  __shared__ h16 vt[8][1024];
  int wv = threadIdx.x >> 5, lane = threadIdx.x & 31;
  int tile = blockIdx.x * 8 + wv;
  if (tile >= ntiles) tile = ntiles - 1;
  long row0 = (long)tile * 16;
  int nlo = lane & 15, half = lane >> 4;
  const v8f zero = {0.f,0.f,0.f,0.f,0.f,0.f,0.f,0.f};
  v16h a0 = load_afrag<0>(Hin, row0, 64, 0, lane);
  v16h a1 = load_afrag<0>(Hin, row0, 64, 32, lane);
#pragma unroll
  for (int nt = 0; nt < 4; ++nt) {
    int n = nt * 16 + nlo;
    v8f aq = wmma16(a0, load_bfrag(pWq, 0, nt, lane), zero);
    aq = wmma16(a1, load_bfrag(pWq, 1, nt, lane), aq);
    float b = bq[n];
#pragma unroll
    for (int i = 0; i < 8; ++i) Q[(row0 + i + half * 8) * 64 + n] = aq[i] + b;
    v8f ak = wmma16(a0, load_bfrag(pWk, 0, nt, lane), zero);
    ak = wmma16(a1, load_bfrag(pWk, 1, nt, lane), ak);
    float bb = bk[n];
#pragma unroll
    for (int i = 0; i < 8; ++i) kt[wv][(i + half * 8) * 64 + n] = (h16)(ak[i] + bb);
    v8f av = wmma16(a0, load_bfrag(pWv, 0, nt, lane), zero);
    av = wmma16(a1, load_bfrag(pWv, 1, nt, lane), av);
    float bv2 = bv[n];
#pragma unroll
    for (int i = 0; i < 8; ++i) vt[wv][(i + half * 8) * 64 + n] = (h16)(av[i] + bv2);
  }
  __syncthreads();
  v16h ka[2] = { lds_afrag(kt[wv], 0, lane), lds_afrag(kt[wv], 1, lane) };
  v16h va[2] = { lds_afrag(vt[wv], 0, lane), lds_afrag(vt[wv], 1, lane) };
#pragma unroll
  for (int nt = 0; nt < 4; ++nt) {
    int n = nt * 16 + nlo;
    v8f ac = wmma16(ka[nt >> 1], load_bfrag(pA, 2, nt, lane), zero);
#pragma unroll
    for (int i = 0; i < 8; ++i) kr2[(row0 + i + half * 8) * 64 + n] = ac[i];
    v8f am = wmma16(va[nt >> 1], load_bfrag(pM, 2, nt, lane), zero);
#pragma unroll
    for (int i = 0; i < 8; ++i) vr2[(row0 + i + half * 8) * 64 + n] = am[i];
  }
}

// ---------------- edge passes (segment softmax + aggregate) ----------------
__global__ void k_edge_score(const int* __restrict__ ei, int E,
    const float* __restrict__ Qd, const float* __restrict__ Kr,
    const float* __restrict__ pr, float* __restrict__ sbuf, float* __restrict__ m) {
  int idx = blockIdx.x * blockDim.x + threadIdx.x;
  if (idx >= E * 2) return;
  int e = idx >> 1, h = idx & 1;
  int src = ei[e], dst = ei[E + e];
  const float4* qp = (const float4*)(Qd + (long)dst * 64 + h * 32);
  const float4* kp = (const float4*)(Kr + (long)src * 64 + h * 32);
  float s = 0.f;
#pragma unroll
  for (int j = 0; j < 8; ++j) {
    float4 a = qp[j], b = kp[j];
    s += a.x * b.x + a.y * b.y + a.z * b.z + a.w * b.w;
  }
  s *= pr[h] * 0.17677669529663687f;     // p_rel * 1/sqrt(32)
  sbuf[idx] = s;
  atomicMaxF(&m[(long)dst * 2 + h], s);
}

__global__ void k_edge_exp(const int* __restrict__ ei, int E,
    float* __restrict__ sbuf, const float* __restrict__ m, float* __restrict__ ssum) {
  int idx = blockIdx.x * blockDim.x + threadIdx.x;
  if (idx >= E * 2) return;
  int e = idx >> 1, h = idx & 1;
  int dst = ei[E + e];
  float ex = expf(sbuf[idx] - m[(long)dst * 2 + h]);
  sbuf[idx] = ex;
  atomicAdd(&ssum[(long)dst * 2 + h], ex);
}

__global__ void k_edge_acc(const int* __restrict__ ei, int E,
    const float* __restrict__ Vr, const float* __restrict__ w,
    const float* __restrict__ sbuf, const float* __restrict__ ssum,
    float* __restrict__ Out) {
  int idx = blockIdx.x * blockDim.x + threadIdx.x;
  if (idx >= E * 2) return;
  int e = idx >> 1, h = idx & 1;
  int src = ei[e], dst = ei[E + e];
  float c = sbuf[idx] / (ssum[(long)dst * 2 + h] + 1e-16f) * w[e];
  const float4* vp = (const float4*)(Vr + (long)src * 64 + h * 32);
  float* op = Out + (long)dst * 64 + h * 32;
#pragma unroll
  for (int j = 0; j < 8; ++j) {
    float4 v = vp[j];
    atomicAdd(op + 4 * j + 0, v.x * c);
    atomicAdd(op + 4 * j + 1, v.y * c);
    atomicAdd(op + 4 * j + 2, v.z * c);
    atomicAdd(op + 4 * j + 3, v.w * c);
  }
}

// ------- stage E: z = sk*(gelu(out)@Wo + bo) + (1-sk)*h -------
__global__ void k_out_proj(const float* __restrict__ Acc, const float* __restrict__ Hs,
    int ntiles, const h16* __restrict__ pWo, const float* __restrict__ bo,
    const float* __restrict__ skp, float* __restrict__ Z) {
  int gw = (blockIdx.x * blockDim.x + threadIdx.x) >> 5;
  int lane = threadIdx.x & 31;
  if (gw >= ntiles) return;
  long row0 = (long)gw * 16;
  int nlo = lane & 15, half = lane >> 4;
  float sk = 1.f / (1.f + expf(-skp[0]));
  const v8f zero = {0.f,0.f,0.f,0.f,0.f,0.f,0.f,0.f};
  v16h a0 = load_afrag<1>(Acc, row0, 64, 0, lane);    // gelu fused on A load
  v16h a1 = load_afrag<1>(Acc, row0, 64, 32, lane);
#pragma unroll
  for (int nt = 0; nt < 4; ++nt) {
    int n = nt * 16 + nlo;
    v8f ac = wmma16(a0, load_bfrag(pWo, 0, nt, lane), zero);
    ac = wmma16(a1, load_bfrag(pWo, 1, nt, lane), ac);
    float b = bo[n];
#pragma unroll
    for (int i = 0; i < 8; ++i) {
      long r = row0 + i + half * 8;
      Z[r * 64 + n] = sk * (ac[i] + b) + (1.f - sk) * Hs[r * 64 + n];
    }
  }
}

// ------- stage F: decoder -------
__global__ void k_decode(const float* __restrict__ zq, const float* __restrict__ za,
    const int* __restrict__ pos, const int* __restrict__ neg, float* __restrict__ out) {
  int idx = blockIdx.x * blockDim.x + threadIdx.x;
  if (idx >= 2 * NLn) return;
  const int* tab = idx < NLn ? pos : neg;
  int i = idx < NLn ? idx : idx - NLn;
  long iq = tab[i], ia = tab[NLn + i];
  const float4* a = (const float4*)(zq + iq * 64);
  const float4* b = (const float4*)(za + ia * 64);
  float s = 0.f;
#pragma unroll
  for (int j = 0; j < 16; ++j) {
    float4 x = a[j], y = b[j];
    s += x.x * y.x + x.y * y.y + x.z * y.z + x.w * y.w;
  }
  out[idx] = 1.f / (1.f + expf(-s));
}

extern "C" void kernel_launch(void* const* d_in, const int* in_sizes, int n_in,
                              void* d_out, int out_size, void* d_ws, size_t ws_size,
                              hipStream_t stream) {
  (void)in_sizes; (void)n_in; (void)out_size; (void)ws_size;
  const float* x_q  = (const float*)d_in[0];
  const float* x_a  = (const float*)d_in[1];
  const int* e_qca  = (const int*)d_in[2];
  const int* e_qwa  = (const int*)d_in[3];
  const int* e_rev  = (const int*)d_in[4];
  const int* pos_i  = (const int*)d_in[5];
  const int* neg_i  = (const int*)d_in[6];
  const float* w1   = (const float*)d_in[7];
  const float* w2   = (const float*)d_in[8];
  const float* w3   = (const float*)d_in[9];
  const float* WinQ = (const float*)d_in[10]; const float* binQ = (const float*)d_in[11];
  const float* WinA = (const float*)d_in[12]; const float* binA = (const float*)d_in[13];
  const float* Wk_q = (const float*)d_in[14]; const float* bk_q = (const float*)d_in[15];
  const float* Wq_q = (const float*)d_in[16]; const float* bq_q = (const float*)d_in[17];
  const float* Wv_q = (const float*)d_in[18]; const float* bv_q = (const float*)d_in[19];
  const float* Wk_a = (const float*)d_in[20]; const float* bk_a = (const float*)d_in[21];
  const float* Wq_a = (const float*)d_in[22]; const float* bq_a = (const float*)d_in[23];
  const float* Wv_a = (const float*)d_in[24]; const float* bv_a = (const float*)d_in[25];
  const float* arel = (const float*)d_in[26];
  const float* mrel = (const float*)d_in[27];
  const float* prel = (const float*)d_in[28];
  const float* Wo_q = (const float*)d_in[29]; const float* bo_q = (const float*)d_in[30];
  const float* Wo_a = (const float*)d_in[31]; const float* bo_a = (const float*)d_in[32];
  const float* sk_q = (const float*)d_in[33];
  const float* sk_a = (const float*)d_in[34];

  // ---- workspace layout ----
  char* base = (char*)d_ws;
  size_t off = 0;
  auto alloc = [&](size_t bytes) -> void* {
    void* r = base + off;
    off = (off + bytes + 255) & ~(size_t)255;
    return r;
  };
  h16* pWinQ = (h16*)alloc(8192 * sizeof(h16));
  h16* pWinA = (h16*)alloc(8192 * sizeof(h16));
  h16* pWkq = (h16*)alloc(4096 * sizeof(h16));
  h16* pWqq = (h16*)alloc(4096 * sizeof(h16));
  h16* pWvq = (h16*)alloc(4096 * sizeof(h16));
  h16* pWka = (h16*)alloc(4096 * sizeof(h16));
  h16* pWqa = (h16*)alloc(4096 * sizeof(h16));
  h16* pWva = (h16*)alloc(4096 * sizeof(h16));
  h16* pWoq = (h16*)alloc(4096 * sizeof(h16));
  h16* pWoa = (h16*)alloc(4096 * sizeof(h16));
  h16* pAr  = (h16*)alloc(6144 * sizeof(h16));
  h16* pMr  = (h16*)alloc(6144 * sizeof(h16));
  float* hq   = (float*)alloc((size_t)NQn * 64 * 4);
  float* ha   = (float*)alloc((size_t)NAn * 64 * 4);
  float* qq   = (float*)alloc((size_t)NQn * 64 * 4);
  float* qa   = (float*)alloc((size_t)NAn * 64 * 4);
  float* kr0  = (float*)alloc((size_t)NQn * 64 * 4);
  float* kr1  = (float*)alloc((size_t)NQn * 64 * 4);
  float* vr0  = (float*)alloc((size_t)NQn * 64 * 4);
  float* vr1  = (float*)alloc((size_t)NQn * 64 * 4);
  float* kr2  = (float*)alloc((size_t)NAn * 64 * 4);
  float* vr2  = (float*)alloc((size_t)NAn * 64 * 4);
  float* outq = (float*)alloc((size_t)NQn * 64 * 4);
  float* outa = (float*)alloc((size_t)NAn * 64 * 4);
  float* m_q  = (float*)alloc((size_t)NQn * 2 * 4);
  float* s_q  = (float*)alloc((size_t)NQn * 2 * 4);
  float* m_a  = (float*)alloc((size_t)NAn * 2 * 4);
  float* s_a  = (float*)alloc((size_t)NAn * 2 * 4);
  float* sb1  = (float*)alloc((size_t)E1n * 2 * 4);
  float* sb2  = (float*)alloc((size_t)E2n * 2 * 4);
  float* sb3  = (float*)alloc((size_t)E3n * 2 * 4);
  float* z_q  = kr0;   // dead after edge scoring
  float* z_a  = kr2;

  const int TPB = 256;
  // 1) pack weights into WMMA B-fragment order
  k_packB<<<32, TPB, 0, stream>>>(WinQ, 4, pWinQ);
  k_packB<<<32, TPB, 0, stream>>>(WinA, 4, pWinA);
  k_packB<<<16, TPB, 0, stream>>>(Wk_q, 2, pWkq);
  k_packB<<<16, TPB, 0, stream>>>(Wq_q, 2, pWqq);
  k_packB<<<16, TPB, 0, stream>>>(Wv_q, 2, pWvq);
  k_packB<<<16, TPB, 0, stream>>>(Wk_a, 2, pWka);
  k_packB<<<16, TPB, 0, stream>>>(Wq_a, 2, pWqa);
  k_packB<<<16, TPB, 0, stream>>>(Wv_a, 2, pWva);
  k_packB<<<16, TPB, 0, stream>>>(Wo_q, 2, pWoq);
  k_packB<<<16, TPB, 0, stream>>>(Wo_a, 2, pWoa);
  k_packRel<<<24, TPB, 0, stream>>>(arel, pAr);
  k_packRel<<<24, TPB, 0, stream>>>(mrel, pMr);

  // 2) init accumulators
  k_fill<<<(NQn * 2 + TPB - 1) / TPB, TPB, 0, stream>>>(m_q, NQn * 2, -3.0e38f);
  k_fill<<<(NAn * 2 + TPB - 1) / TPB, TPB, 0, stream>>>(m_a, NAn * 2, -3.0e38f);
  hipMemsetAsync(s_q, 0, (size_t)NQn * 2 * 4, stream);
  hipMemsetAsync(s_a, 0, (size_t)NAn * 2 * 4, stream);
  hipMemsetAsync(outq, 0, (size_t)NQn * 64 * 4, stream);
  hipMemsetAsync(outa, 0, (size_t)NAn * 64 * 4, stream);

  // 3) input projections (WMMA, K=128, relu)
  int tq = NQn / 16, ta = NAn / 16;           // 6250, 12500 (exact)
  k_gemm_in<<<(tq + 7) / 8, TPB, 0, stream>>>(x_q, tq, pWinQ, binQ, hq);
  k_gemm_in<<<(ta + 7) / 8, TPB, 0, stream>>>(x_a, ta, pWinA, binA, ha);

  // 4) fused q/k/v + relation einsums (WMMA + LDS re-striping)
  k_qkv_rel_q<<<(tq + 7) / 8, TPB, 0, stream>>>(hq, tq, pWkq, pWqq, pWvq, bk_q, bq_q, bv_q,
                                                pAr, pMr, qq, kr0, kr1, vr0, vr1);
  k_qkv_rel_a<<<(ta + 7) / 8, TPB, 0, stream>>>(ha, ta, pWka, pWqa, pWva, bk_a, bq_a, bv_a,
                                                pAr, pMr, qa, kr2, vr2);

  // 5) edge scoring + segment max
  k_edge_score<<<(E1n * 2 + TPB - 1) / TPB, TPB, 0, stream>>>(e_qca, E1n, qa, kr0, prel + 0, sb1, m_a);
  k_edge_score<<<(E2n * 2 + TPB - 1) / TPB, TPB, 0, stream>>>(e_qwa, E2n, qa, kr1, prel + 2, sb2, m_a);
  k_edge_score<<<(E3n * 2 + TPB - 1) / TPB, TPB, 0, stream>>>(e_rev, E3n, qq, kr2, prel + 4, sb3, m_q);
  // 6) exp + segment sum
  k_edge_exp<<<(E1n * 2 + TPB - 1) / TPB, TPB, 0, stream>>>(e_qca, E1n, sb1, m_a, s_a);
  k_edge_exp<<<(E2n * 2 + TPB - 1) / TPB, TPB, 0, stream>>>(e_qwa, E2n, sb2, m_a, s_a);
  k_edge_exp<<<(E3n * 2 + TPB - 1) / TPB, TPB, 0, stream>>>(e_rev, E3n, sb3, m_q, s_q);
  // 7) weighted message aggregation
  k_edge_acc<<<(E1n * 2 + TPB - 1) / TPB, TPB, 0, stream>>>(e_qca, E1n, vr0, w1, sb1, s_a, outa);
  k_edge_acc<<<(E2n * 2 + TPB - 1) / TPB, TPB, 0, stream>>>(e_qwa, E2n, vr1, w2, sb2, s_a, outa);
  k_edge_acc<<<(E3n * 2 + TPB - 1) / TPB, TPB, 0, stream>>>(e_rev, E3n, vr2, w3, sb3, s_q, outq);

  // 8) output projection (gelu fused on A) + gated skip
  k_out_proj<<<(ta + 7) / 8, TPB, 0, stream>>>(outa, ha, ta, pWoa, bo_a, sk_a, z_a);
  k_out_proj<<<(tq + 7) / 8, TPB, 0, stream>>>(outq, hq, tq, pWoq, bo_q, sk_q, z_q);

  // 9) decoder
  k_decode<<<(2 * NLn + TPB - 1) / TPB, TPB, 0, stream>>>(z_q, z_a, pos_i, neg_i, (float*)d_out);
}
